// DualModel_75651553952271
// MI455X (gfx1250) — matrix operand order repaired and checked by reading, compiled, not run on hardware
//
#include <hip/hip_runtime.h>
#include <hip/hip_bf16.h>

typedef __attribute__((ext_vector_type(16))) _Float16 v16h;
typedef __attribute__((ext_vector_type(4)))  _Float16 h4;
typedef __attribute__((ext_vector_type(8)))  float    v8f;

#define NNODE 1200
#define EEDGE 719400           // N*(N-1)/2, complete graph
#define SDIM  4800
#define H_ELEMS (NNODE * 64)   // 76800
#define WAVES 8
#define TPW   4                // edge tiles per wave
#define SLOPE 0.1f

#define WMMA32(A, B, C) \
    __builtin_amdgcn_wmma_f32_16x16x32_f16(false, (A), false, (B), (short)0, (C), false, false)

__constant__ int c_smat[16] = {0,1,2,3,1,4,5,6,2,5,7,8,3,6,8,9};

// 16-bit A-fragment LDS layout used here (contiguous k = 16*(lane/16) + s, same as B): element (row m, k) lives at
//   [ (k>>5)*512 + (m + 16*((k>>4)&1))*16 + (k&15) ]
// so lane L's v16h fragment for chunk kc is the contiguous 32B at [kc*512 + L*16].
__device__ __forceinline__ int aidx(int m, int k) {
    return ((k >> 5) << 9) + ((m + (((k >> 4) & 1) << 4)) << 4) + (k & 15);
}

// ---------------------------------------------------------------------------
// Collapsed-GCN helpers (complete graph: agg = (colsum(x)-x_row)@W / (N-1) + b)
// ---------------------------------------------------------------------------
__global__ void colsum_kernel(const float* __restrict__ in, int rows, int stride,
                              int cols, float* __restrict__ out) {
    int c = blockIdx.x * blockDim.x + threadIdx.x;
    if (c >= cols) return;
    float s = 0.f;
    for (int r = 0; r < rows; ++r) s += in[r * stride + c];
    out[c] = s;
}

__global__ void gcn_layer_kernel(const float* __restrict__ in, int stride, int din,
                                 const float* __restrict__ colsum,
                                 const float* __restrict__ W, const float* __restrict__ b,
                                 int dout, float* __restrict__ out, int outstride,
                                 int leaky) {
    int row = blockIdx.x * blockDim.x + threadIdx.x;
    if (row >= NNODE) return;
    float xr[32];
    for (int k = 0; k < din; ++k) xr[k] = colsum[k] - in[row * stride + k];
    const float inv = 1.0f / (float)(NNODE - 1);
    for (int c = 0; c < dout; ++c) {
        float acc = 0.f;
        for (int k = 0; k < din; ++k) acc += xr[k] * W[k * dout + c];
        float v = acc * inv + b[c];
        if (leaky) v = fmaxf(v, SLOPE * v);     // slope<1: leaky == max(v, s*v)
        out[row * outstride + c] = v;
    }
}

// ---------------------------------------------------------------------------
// Edge MLP + off-diagonal S scatter. One wave handles TPW 16-edge tiles.
// ---------------------------------------------------------------------------
__global__ void __launch_bounds__(256)
edge_kernel(const float* __restrict__ h, const int* __restrict__ ud,
            const float* __restrict__ We1, const float* __restrict__ be1,
            const float* __restrict__ We2, const float* __restrict__ be2,
            const float* __restrict__ We3, const float* __restrict__ be3,
            float* __restrict__ S) {
    __shared__ __align__(32) _Float16 wb1[4096];   // [nt][kc][lane][q] B-fragments
    __shared__ __align__(32) _Float16 wb2[4096];
    __shared__ __align__(32) _Float16 wb3[1024];
    __shared__ float b1s[64], b2s[64], b3s[16];
    __shared__ __align__(32) _Float16 ab[WAVES][1024];
    __shared__ __align__(32) _Float16 bb[WAVES][1024];
    __shared__ int eij[WAVES][32];

    const int t = threadIdx.x;
    // Stage weights pre-swizzled into B-fragment layout:
    // dst[((nt*2+kc)*32+ln)*16+q] = W[(kc*32 + 16*(ln>=16) + q)*NOUT + nt*16 + (ln&15)]
    for (int i = t; i < 4096; i += 256) {
        const int q = i & 15, ln = (i >> 4) & 31, kc = (i >> 9) & 1, nt = i >> 10;
        const int k = kc * 32 + ((ln & 16) ? 16 : 0) + q;
        const int n = nt * 16 + (ln & 15);
        wb1[i] = (_Float16)We1[k * 64 + n];
        wb2[i] = (_Float16)We2[k * 64 + n];
    }
    for (int i = t; i < 1024; i += 256) {
        const int q = i & 15, ln = (i >> 4) & 31, kc = (i >> 9) & 1;
        const int k = kc * 32 + ((ln & 16) ? 16 : 0) + q;
        wb3[i] = (_Float16)We3[k * 16 + (ln & 15)];
    }
    if (t < 64) { b1s[t] = be1[t]; b2s[t] = be2[t]; }
    if (t < 16) b3s[t] = be3[t];
    __syncthreads();

    const int wave = t >> 5, lane = t & 31;
    const bool hi = lane >= 16;
    const int n15 = lane & 15;
    const long tile0 = ((long)blockIdx.x * WAVES + wave) * TPW;

    for (int tt = 0; tt < TPW; ++tt) {
        const long ebase = (tile0 + tt) * 16L;
        if (ebase >= EEDGE) break;   // wave-uniform: full EXEC maintained

        if (lane < 16) {
            long e = ebase + lane; if (e >= EEDGE) e = EEDGE - 1;
            eij[wave][lane]      = ud[2 * e];
            eij[wave][lane + 16] = ud[2 * e + 1];
        }

        // Build he tile (16 edges x 64 feats) straight into A-fragment layout
        for (int idx = lane; idx < 256; idx += 32) {
            const int m = idx >> 4, k0 = (idx & 15) << 2;
            const int i = eij[wave][m], j = eij[wave][m + 16];
            const float4 va = *(const float4*)(h + i * 64 + k0);
            const float4 vb = *(const float4*)(h + j * 64 + k0);
            h4 v;
            v[0] = (_Float16)(va.x + vb.x); v[1] = (_Float16)(va.y + vb.y);
            v[2] = (_Float16)(va.z + vb.z); v[3] = (_Float16)(va.w + vb.w);
            *(h4*)&ab[wave][aidx(m, k0)] = v;
        }

        // ---- layer 1: ab -> bb (leaky) ----
        {
            v16h a0 = *(const v16h*)&ab[wave][lane * 16];
            v16h a1 = *(const v16h*)&ab[wave][512 + lane * 16];
#pragma unroll
            for (int nt = 0; nt < 4; ++nt) {
                v16h q0 = *(const v16h*)&wb1[(nt * 2 + 0) * 512 + lane * 16];
                v16h q1 = *(const v16h*)&wb1[(nt * 2 + 1) * 512 + lane * 16];
                const int n = nt * 16 + n15;
                const float bn = b1s[n];
                v8f c = {bn, bn, bn, bn, bn, bn, bn, bn};   // bias folded into C
                c = WMMA32(a0, q0, c);
                c = WMMA32(a1, q1, c);
#pragma unroll
                for (int r = 0; r < 8; ++r) {
                    const int M = r + (hi ? 8 : 0);
                    const float v = fmaxf(c[r], SLOPE * c[r]);
                    bb[wave][aidx(M, n)] = (_Float16)v;
                }
            }
        }
        // ---- layer 2: bb -> ab (leaky) ----
        {
            v16h a0 = *(const v16h*)&bb[wave][lane * 16];
            v16h a1 = *(const v16h*)&bb[wave][512 + lane * 16];
#pragma unroll
            for (int nt = 0; nt < 4; ++nt) {
                v16h q0 = *(const v16h*)&wb2[(nt * 2 + 0) * 512 + lane * 16];
                v16h q1 = *(const v16h*)&wb2[(nt * 2 + 1) * 512 + lane * 16];
                const int n = nt * 16 + n15;
                const float bn = b2s[n];
                v8f c = {bn, bn, bn, bn, bn, bn, bn, bn};
                c = WMMA32(a0, q0, c);
                c = WMMA32(a1, q1, c);
#pragma unroll
                for (int r = 0; r < 8; ++r) {
                    const int M = r + (hi ? 8 : 0);
                    const float v = fmaxf(c[r], SLOPE * c[r]);
                    ab[wave][aidx(M, n)] = (_Float16)v;
                }
            }
        }
        // ---- layer 3: ab -> 16x16 accumulator (bias in C) ----
        const float bias3 = b3s[n15];
        v8f c = {bias3, bias3, bias3, bias3, bias3, bias3, bias3, bias3};
        {
            v16h a0 = *(const v16h*)&ab[wave][lane * 16];
            v16h a1 = *(const v16h*)&ab[wave][512 + lane * 16];
            v16h q0 = *(const v16h*)&wb3[lane * 16];
            v16h q1 = *(const v16h*)&wb3[512 + lane * 16];
            c = WMMA32(a0, q0, c);
            c = WMMA32(a1, q1, c);
        }
        // Scatter 4x4 blocks: S[4i+a][4j+b] = ed[e][4a+b]; transposed at (j,i)
        const int a4 = n15 >> 2, b4 = n15 & 3;
#pragma unroll
        for (int r = 0; r < 8; ++r) {
            const int M = r + (hi ? 8 : 0);
            if (ebase + M >= EEDGE) continue;
            const int i = eij[wave][M], j = eij[wave][M + 16];
            S[(4 * i + a4) * SDIM + (4 * j + b4)] = c[r];
            S[(4 * j + b4) * SDIM + (4 * i + a4)] = c[r];
        }
    }
}

// ---------------------------------------------------------------------------
// Node MLP + diagonal (Vsym) S scatter. One wave per 16-node tile.
// ---------------------------------------------------------------------------
__global__ void __launch_bounds__(256)
node_kernel(const float* __restrict__ h,
            const float* __restrict__ Wn1, const float* __restrict__ bn1,
            const float* __restrict__ Wn2, const float* __restrict__ bn2,
            const float* __restrict__ Wn3, const float* __restrict__ bn3,
            float* __restrict__ S) {
    __shared__ __align__(32) _Float16 wb1[4096];
    __shared__ __align__(32) _Float16 wb2[4096];
    __shared__ __align__(32) _Float16 wb3[1024];
    __shared__ float b1s[64], b2s[64], b3s[16];
    __shared__ __align__(32) _Float16 ab[WAVES][1024];
    __shared__ __align__(32) _Float16 bb[WAVES][1024];

    const int t = threadIdx.x;
    for (int i = t; i < 4096; i += 256) {
        const int q = i & 15, ln = (i >> 4) & 31, kc = (i >> 9) & 1, nt = i >> 10;
        const int k = kc * 32 + ((ln & 16) ? 16 : 0) + q;
        const int n = nt * 16 + (ln & 15);
        wb1[i] = (_Float16)Wn1[k * 64 + n];
        wb2[i] = (_Float16)Wn2[k * 64 + n];
    }
    for (int i = t; i < 1024; i += 256) {        // pad 64x10 -> 64x16 with zeros
        const int q = i & 15, ln = (i >> 4) & 31, kc = (i >> 9) & 1;
        const int k = kc * 32 + ((ln & 16) ? 16 : 0) + q;
        const int n = ln & 15;
        wb3[i] = (n < 10) ? (_Float16)Wn3[k * 10 + n] : (_Float16)0.0f;
    }
    if (t < 64) { b1s[t] = bn1[t]; b2s[t] = bn2[t]; }
    if (t < 16) b3s[t] = (t < 10) ? bn3[t] : 0.f;
    __syncthreads();

    const int wave = t >> 5, lane = t & 31;
    const bool hi = lane >= 16;
    const int n15 = lane & 15;
    const int nbase = (blockIdx.x * WAVES + wave) * 16;
    if (nbase >= NNODE) return;

    for (int idx = lane; idx < 256; idx += 32) {
        const int m = idx >> 4, k0 = (idx & 15) << 2;
        const float4 va = *(const float4*)(h + (nbase + m) * 64 + k0);
        h4 v;
        v[0] = (_Float16)va.x; v[1] = (_Float16)va.y;
        v[2] = (_Float16)va.z; v[3] = (_Float16)va.w;
        *(h4*)&ab[wave][aidx(m, k0)] = v;
    }

    {
        v16h a0 = *(const v16h*)&ab[wave][lane * 16];
        v16h a1 = *(const v16h*)&ab[wave][512 + lane * 16];
#pragma unroll
        for (int nt = 0; nt < 4; ++nt) {
            v16h q0 = *(const v16h*)&wb1[(nt * 2 + 0) * 512 + lane * 16];
            v16h q1 = *(const v16h*)&wb1[(nt * 2 + 1) * 512 + lane * 16];
            const int n = nt * 16 + n15;
            const float bn = b1s[n];
            v8f c = {bn, bn, bn, bn, bn, bn, bn, bn};
            c = WMMA32(a0, q0, c);
            c = WMMA32(a1, q1, c);
#pragma unroll
            for (int r = 0; r < 8; ++r) {
                const int M = r + (hi ? 8 : 0);
                const float v = fmaxf(c[r], SLOPE * c[r]);
                bb[wave][aidx(M, n)] = (_Float16)v;
            }
        }
    }
    {
        v16h a0 = *(const v16h*)&bb[wave][lane * 16];
        v16h a1 = *(const v16h*)&bb[wave][512 + lane * 16];
#pragma unroll
        for (int nt = 0; nt < 4; ++nt) {
            v16h q0 = *(const v16h*)&wb2[(nt * 2 + 0) * 512 + lane * 16];
            v16h q1 = *(const v16h*)&wb2[(nt * 2 + 1) * 512 + lane * 16];
            const int n = nt * 16 + n15;
            const float bn = b2s[n];
            v8f c = {bn, bn, bn, bn, bn, bn, bn, bn};
            c = WMMA32(a0, q0, c);
            c = WMMA32(a1, q1, c);
#pragma unroll
            for (int r = 0; r < 8; ++r) {
                const int M = r + (hi ? 8 : 0);
                const float v = fmaxf(c[r], SLOPE * c[r]);
                ab[wave][aidx(M, n)] = (_Float16)v;
            }
        }
    }
    const float bias3 = b3s[n15];
    v8f c = {bias3, bias3, bias3, bias3, bias3, bias3, bias3, bias3};
    {
        v16h a0 = *(const v16h*)&ab[wave][lane * 16];
        v16h a1 = *(const v16h*)&ab[wave][512 + lane * 16];
        v16h q0 = *(const v16h*)&wb3[lane * 16];
        v16h q1 = *(const v16h*)&wb3[512 + lane * 16];
        c = WMMA32(a0, q0, c);
        c = WMMA32(a1, q1, c);
    }
    // Park vd tile as f32 (reuse bb wave slot: 1KB <= 2KB), then scatter Vsym
    float* vb = reinterpret_cast<float*>(&bb[wave][0]);
#pragma unroll
    for (int r = 0; r < 8; ++r) {
        const int M = r + (hi ? 8 : 0);
        vb[M * 16 + n15] = c[r];
    }
    for (int idx = lane; idx < 256; idx += 32) {
        const int M = idx >> 4, p = idx & 15;
        const int node = nbase + M;
        const float val = vb[M * 16 + c_smat[p]];
        const int a = p >> 2, b = p & 3;
        S[(4 * node + a) * SDIM + (4 * node + b)] = val;
    }
}

// ---------------------------------------------------------------------------
extern "C" void kernel_launch(void* const* d_in, const int* in_sizes, int n_in,
                              void* d_out, int out_size, void* d_ws, size_t ws_size,
                              hipStream_t stream) {
    const float* x   = (const float*)d_in[0];   // (1200,16) -> use cols 0..5
    const int*   ud  = (const int*)  d_in[3];   // (E,2)
    const float* Wg1 = (const float*)d_in[4];  const float* bg1 = (const float*)d_in[5];
    const float* Wg2 = (const float*)d_in[6];  const float* bg2 = (const float*)d_in[7];
    const float* Wg3 = (const float*)d_in[8];  const float* bg3 = (const float*)d_in[9];
    const float* Wn1 = (const float*)d_in[10]; const float* bn1 = (const float*)d_in[11];
    const float* Wn2 = (const float*)d_in[12]; const float* bn2 = (const float*)d_in[13];
    const float* Wn3 = (const float*)d_in[14]; const float* bn3 = (const float*)d_in[15];
    const float* We1 = (const float*)d_in[16]; const float* be1 = (const float*)d_in[17];
    const float* We2 = (const float*)d_in[18]; const float* be2 = (const float*)d_in[19];
    const float* We3 = (const float*)d_in[20]; const float* be3 = (const float*)d_in[21];

    float* out   = (float*)d_out;
    float* h_out = out;               // (1200,64)
    float* S     = out + H_ELEMS;     // (4800,4800)

    float* ws = (float*)d_ws;
    float* h1 = ws;                   // 1200*32
    float* h2 = ws + NNODE * 32;      // 1200*32
    float* cs = ws + NNODE * 64;      // up to 64 colsums

    const int rb = (NNODE + 255) / 256;

    colsum_kernel<<<1, 64, 0, stream>>>(x, NNODE, 16, 6, cs);
    gcn_layer_kernel<<<rb, 256, 0, stream>>>(x, 16, 6, cs, Wg1, bg1, 32, h1, 32, 1);
    colsum_kernel<<<1, 64, 0, stream>>>(h1, NNODE, 32, 32, cs);
    gcn_layer_kernel<<<rb, 256, 0, stream>>>(h1, 32, 32, cs, Wg2, bg2, 32, h2, 32, 1);
    colsum_kernel<<<1, 64, 0, stream>>>(h2, NNODE, 32, 32, cs);
    gcn_layer_kernel<<<rb, 256, 0, stream>>>(h2, 32, 32, cs, Wg3, bg3, 64, h_out, 64, 0);

    node_kernel<<<(75 + WAVES - 1) / WAVES, 256, 0, stream>>>(h_out, Wn1, bn1, Wn2, bn2,
                                                              Wn3, bn3, S);
    const int etiles = (EEDGE + 15) / 16;                       // 44963
    const int eblocks = (etiles + WAVES * TPW - 1) / (WAVES * TPW);
    edge_kernel<<<eblocks, 256, 0, stream>>>(h_out, ud, We1, be1, We2, be2, We3, be3, S);
}
